// GCNFoldImproved_30897994727576
// MI455X (gfx1250) — compile-verified, matrix operand order checked by reading, not run on hardware
//
#include <hip/hip_runtime.h>

typedef __attribute__((ext_vector_type(16))) _Float16 v16h;
typedef __attribute__((ext_vector_type(8)))  float    v8f;
typedef __attribute__((ext_vector_type(4)))  unsigned uint32x4;
typedef __attribute__((ext_vector_type(8)))  int      int32x8;
typedef __attribute__((ext_vector_type(4)))  int      int32x4;

#define S_LEN 256
#define H_DIM 256

#if defined(__has_builtin)
#if __has_builtin(__builtin_amdgcn_tensor_load_to_lds) && __has_builtin(__builtin_amdgcn_s_wait_tensorcnt)
#define USE_TDM 1
#endif
#endif

// Even-half K base inside a 16x32 f16 WMMA fragment (CDNA5 ISA 7.12.2 A layout;
// B mirrored: lane -> column). Fragment dword v covers K = kbase, kbase+1.
__device__ __forceinline__ int kbase16(int v, int lanehi) {
    return ((v & 4) ? 16 : 0) + (lanehi ? 8 : 0) + ((v & 3) << 1);
}

// Load a v16h fragment from one padded LDS row as 8 dword reads.
__device__ __forceinline__ v16h load_frag_row(const _Float16* row, int lanehi) {
    union { v16h f; unsigned u[8]; } r;
#pragma unroll
    for (int v = 0; v < 8; ++v)
        r.u[v] = *(const unsigned*)(row + kbase16(v, lanehi));
    return r.f;
}

#ifdef USE_TDM
// TDM 2D tile load: global (tile_d1 rows x tile_d0 f32, row stride stride0)
// -> LDS at lds_off, inserting 4 pad dwords after every 256 dwords
// (destination rows become 260 dwords; bank-conflict-free for column reads).
__device__ __forceinline__ void tdm_load_tile_f32(unsigned lds_off, const float* gptr,
                                                  int tensor_d0, int tensor_d1,
                                                  int stride0, int tile_d0, int tile_d1)
{
    unsigned long long ga = (unsigned long long)(uintptr_t)gptr;
    uint32x4 g0;
    g0[0] = 1u;                                   // count=1, user mode, no gather
    g0[1] = lds_off;                              // LDS byte address
    g0[2] = (unsigned)(ga & 0xffffffffu);         // global_addr[31:0]
    g0[3] = (unsigned)((ga >> 32) & 0x1ffffffu) | (2u << 30);  // addr[56:32] | type=2
    int32x8 g1;
    // data_size=4B | pad_enable | pad_interval=256dw | pad_amount=4dw
    g1[0] = (int)((2u << 16) | (1u << 20) | (7u << 22) | (3u << 25));
    g1[1] = (int)(((unsigned)tensor_d0 & 0xffffu) << 16);
    g1[2] = (int)((((unsigned)tensor_d0 >> 16) & 0xffffu) |
                  (((unsigned)tensor_d1 & 0xffffu) << 16));
    g1[3] = (int)((((unsigned)tensor_d1 >> 16) & 0xffffu) |
                  (((unsigned)tile_d0 & 0xffffu) << 16));
    g1[4] = (int)((unsigned)tile_d1 & 0xffffu);   // tile_dim1 | tile_dim2=0
    g1[5] = stride0;                              // tensor_dim0_stride[31:0]
    g1[6] = 0;
    g1[7] = 0;
    int32x4 z4 = {0, 0, 0, 0};
#if __has_include(<hip/amd_detail/amd_gfx1250_TDM.h>)
    int32x8 z8 = {0, 0, 0, 0, 0, 0, 0, 0};
    __builtin_amdgcn_tensor_load_to_lds(g0, g1, z4, z4, z8, 0);   // clang-23 arity
#else
    __builtin_amdgcn_tensor_load_to_lds(g0, g1, z4, z4, 0);       // ROCm 7.2 arity
#endif
}
#endif

__global__ __launch_bounds__(256)
void zero_fill(float* __restrict__ p, int n) {
    int i = blockIdx.x * 256 + threadIdx.x;
    if (i < n) p[i] = 0.f;
}

// Generic C[b] = act(A[b] @ B[b] + bias), fp32 storage, f16 WMMA fragments.
// Block = 256 threads (8 waves) computes a 128x16 tile; panels staged in LDS
// with coalesced loads; fragments read as dword LDS loads.
// Grid: (N/16, M/128, batch). M, K multiples of 128 / 32.
template <int RELU>
__global__ __launch_bounds__(256)
void gemm_f32_wmma(const float* __restrict__ A, int lda, long long sA,
                   const float* __restrict__ Bw, int ldb, long long sB,
                   const float* __restrict__ bias,
                   float* __restrict__ C, int ldc, long long sC, int K)
{
    __shared__ _Float16 As[128][34];   // padded rows: 17 dwords -> conflict-free
    __shared__ _Float16 BsT[16][34];
    const int tid = threadIdx.x, lane = tid & 31, wave = tid >> 5;
    const int m0 = blockIdx.y * 128, n0 = blockIdx.x * 16;
    const long long bz = blockIdx.z;
    A  += bz * sA;
    Bw += bz * sB;
    C  += bz * sC;
    const int lanehi = (lane >= 16) ? 1 : 0;

    v8f acc = {0.f, 0.f, 0.f, 0.f, 0.f, 0.f, 0.f, 0.f};
    for (int k0 = 0; k0 < K; k0 += 32) {
        // Stage A panel 128x32 (coalesced 128B rows).
#pragma unroll 4
        for (int idx = tid; idx < 128 * 32; idx += 256) {
            int r = idx >> 5, k = idx & 31;
            As[r][k] = (_Float16)A[(long long)(m0 + r) * lda + (k0 + k)];
        }
        // Stage B panel transposed 16x32 (coalesced over n).
#pragma unroll
        for (int idx = tid; idx < 32 * 16; idx += 256) {
            int k = idx >> 4, n = idx & 15;
            BsT[n][k] = (_Float16)Bw[(long long)(k0 + k) * ldb + (n0 + n)];
        }
        __syncthreads();
        v16h a = load_frag_row(&As[wave * 16 + (lane & 15)][0], lanehi);
        v16h b = load_frag_row(&BsT[lane & 15][0], lanehi);
        acc = __builtin_amdgcn_wmma_f32_16x16x32_f16(
            false, a, false, b, (short)0, acc, false, false);
        __syncthreads();
    }
    const int n = n0 + (lane & 15);
    const float bb = bias[n];
#pragma unroll
    for (int r = 0; r < 8; ++r) {
        float v = acc[r] + bb;
        if (RELU) v = fmaxf(v, 0.f);
        C[(long long)(m0 + wave * 16 + r + lanehi * 8) * ldc + n] = v;
    }
}

// h = LayerNorm(relu(s + nb)) * g + b, one 256-thread block per row (H=256).
__global__ __launch_bounds__(256)
void relu_add_ln(const float* __restrict__ s, const float* __restrict__ nb,
                 const float* __restrict__ g, const float* __restrict__ b,
                 float* __restrict__ out)
{
    const int row = blockIdx.x, c = threadIdx.x;
    __shared__ float red[256];
    __shared__ float mu_s, var_s;
    float x = fmaxf(s[row * 256 + c] + nb[row * 256 + c], 0.f);
    red[c] = x; __syncthreads();
    for (int off = 128; off > 0; off >>= 1) {
        if (c < off) red[c] += red[c + off];
        __syncthreads();
    }
    if (c == 0) mu_s = red[0] * (1.f / 256.f);
    __syncthreads();
    const float d = x - mu_s;
    red[c] = d * d; __syncthreads();
    for (int off = 128; off > 0; off >>= 1) {
        if (c < off) red[c] += red[c + off];
        __syncthreads();
    }
    if (c == 0) var_s = red[0] * (1.f / 256.f);
    __syncthreads();
    out[row * 256 + c] = d * rsqrtf(var_s + 1e-5f) * g[c] + b[c];
}

// Fused pairwise scorer. Block (8 waves) owns one (b, 16x16 (i,j)) output
// block: 256 pairs (M), K=256, N=128. hi/hj tiles staged once by the TDM,
// V and W2 panels rebuilt per 32-wide K chunk, N reduced in-register.
__global__ __launch_bounds__(256)
void pairwise_score_wmma(const float* __restrict__ hi, const float* __restrict__ hj,
                         const float* __restrict__ bp,
                         const float* __restrict__ w1,   // [513,256] (row 512 = prior wt)
                         const float* __restrict__ b1,   // [256]
                         const float* __restrict__ w2,   // [256,128]
                         const float* __restrict__ b2,   // [128]
                         const float* __restrict__ w3,   // [128]
                         const float* __restrict__ b3,   // [1]
                         float* __restrict__ out)        // [B,S,S]
{
    __shared__ float    hi_s[16][260];   // TDM pad: 256 dw + 4 dw per row
    __shared__ float    hj_s[16][260];
    __shared__ _Float16 Va[256][34];     // activated V panel (pairs x kchunk)
    __shared__ _Float16 BpT[128][34];    // W2 chunk transposed (n x k)
    __shared__ float    prior_s[256];
    __shared__ float    w1r_s[256], b1_s[256];
    __shared__ float    b2_s[128], w3_s[128];

    const int tid  = threadIdx.x;
    const int lane = tid & 31, wave = tid >> 5;
    const int j0 = blockIdx.x * 16, i0 = blockIdx.y * 16;
    const long long bz = blockIdx.z;

    const float* hi_g = hi + (bz * S_LEN + i0) * H_DIM;   // contiguous 16x256 tile
    const float* hj_g = hj + (bz * S_LEN + j0) * H_DIM;

#ifdef USE_TDM
    if (wave == 0) {
        if (lane == 0) {
            tdm_load_tile_f32((unsigned)(uintptr_t)&hi_s[0][0], hi_g,
                              H_DIM, 16, H_DIM, H_DIM, 16);
            tdm_load_tile_f32((unsigned)(uintptr_t)&hj_s[0][0], hj_g,
                              H_DIM, 16, H_DIM, H_DIM, 16);
        }
        __builtin_amdgcn_s_wait_tensorcnt(0);
    }
#else
    for (int idx = tid; idx < 16 * 256; idx += 256) {
        int r = idx >> 8, k = idx & 255;
        hi_s[r][k] = hi_g[r * H_DIM + k];
        hj_s[r][k] = hj_g[r * H_DIM + k];
    }
#endif
    prior_s[tid] = bp[(bz * S_LEN + (i0 + (tid >> 4))) * S_LEN + (j0 + (tid & 15))];
    w1r_s[tid]   = w1[512 * 256 + tid];
    b1_s[tid]    = b1[tid];
    if (tid < 128) { b2_s[tid] = b2[tid]; w3_s[tid] = w3[tid]; }
    __syncthreads();

    const int lanehi = (lane >= 16) ? 1 : 0;
    v8f acc[2][8];
    const v8f zero = {0.f, 0.f, 0.f, 0.f, 0.f, 0.f, 0.f, 0.f};
#pragma unroll
    for (int mt = 0; mt < 2; ++mt)
#pragma unroll
        for (int nt = 0; nt < 8; ++nt) acc[mt][nt] = zero;

    // Thread owns pair p = tid: i_local = p/16, j_local = p%16.
    const int il = tid >> 4, jl = tid & 15;
    const float pr = prior_s[tid];

    for (int kc = 0; kc < 8; ++kc) {
        const int k0 = kc * 32;
        // Activated V panel: relu(hi + hj + prior*w1r + b1) -> f16.
#pragma unroll
        for (int k = 0; k < 32; ++k) {
            float v = hi_s[il][k0 + k] + hj_s[jl][k0 + k]
                    + pr * w1r_s[k0 + k] + b1_s[k0 + k];
            Va[tid][k] = (_Float16)fmaxf(v, 0.f);
        }
        // W2 chunk transposed so K is contiguous per n-row.
#pragma unroll
        for (int idx = tid; idx < 128 * 32; idx += 256) {
            int n = idx >> 5, k = idx & 31;
            BpT[n][k] = (_Float16)w2[(long long)(k0 + k) * 128 + n];
        }
        __syncthreads();

        v16h afr[2];
#pragma unroll
        for (int mt = 0; mt < 2; ++mt)
            afr[mt] = load_frag_row(&Va[(wave * 2 + mt) * 16 + (lane & 15)][0], lanehi);
#pragma unroll
        for (int nt = 0; nt < 8; ++nt) {
            v16h bfr = load_frag_row(&BpT[nt * 16 + (lane & 15)][0], lanehi);
#pragma unroll
            for (int mt = 0; mt < 2; ++mt)
                acc[mt][nt] = __builtin_amdgcn_wmma_f32_16x16x32_f16(
                    false, afr[mt], false, bfr, (short)0, acc[mt][nt], false, false);
        }
        __syncthreads();
    }

    // Epilogue: score[m] = sum_n relu(acc[m,n] + b2[n]) * w3[n] + b3.
    const float bias3 = b3[0];
#pragma unroll
    for (int mt = 0; mt < 2; ++mt) {
        float part[8];
#pragma unroll
        for (int r = 0; r < 8; ++r) part[r] = 0.f;
#pragma unroll
        for (int nt = 0; nt < 8; ++nt) {
            int n = nt * 16 + (lane & 15);
            float bb = b2_s[n], ww = w3_s[n];
#pragma unroll
            for (int r = 0; r < 8; ++r)
                part[r] += fmaxf(acc[mt][nt][r] + bb, 0.f) * ww;
        }
        // Butterfly over the 16 N-columns held per half-wave (wave32).
#pragma unroll
        for (int mask = 1; mask < 16; mask <<= 1)
#pragma unroll
            for (int r = 0; r < 8; ++r)
                part[r] += __shfl_xor(part[r], mask, 32);
        const int t = wave * 2 + mt;   // i_local of this M-tile
        const long long base = (bz * S_LEN + (i0 + t)) * S_LEN + j0;
        if ((lane & 15) == 0) {
            int mbase = lanehi * 8;    // lane0 -> rows 0..7, lane16 -> rows 8..15
#pragma unroll
            for (int r = 0; r < 8; ++r)
                out[base + mbase + r] = part[r] + bias3;
        }
    }
}

extern "C" void kernel_launch(void* const* d_in, const int* in_sizes, int n_in,
                              void* d_out, int out_size, void* d_ws, size_t ws_size,
                              hipStream_t stream)
{
    const float* emb   = (const float*)d_in[0];   // [4,256,640]
    const float* bp    = (const float*)d_in[1];   // [4,256,256]
    // d_in[2]: sequence_lengths (all == S) — unused
    const float* ew1   = (const float*)d_in[3];   // [640,512]
    const float* eb1   = (const float*)d_in[4];
    const float* ew2   = (const float*)d_in[5];   // [512,256]
    const float* eb2   = (const float*)d_in[6];
    const float* wself = (const float*)d_in[7];   // [3,256,256]
    const float* bself = (const float*)d_in[8];
    const float* wedge = (const float*)d_in[9];
    const float* bedge = (const float*)d_in[10];
    const float* lng   = (const float*)d_in[11];
    const float* lnb   = (const float*)d_in[12];
    const float* mw1   = (const float*)d_in[13];  // [513,256]
    const float* mb1   = (const float*)d_in[14];
    const float* mw2   = (const float*)d_in[15];  // [256,128]
    const float* mb2   = (const float*)d_in[16];
    const float* mw3   = (const float*)d_in[17];  // [128,1]
    const float* mb3   = (const float*)d_in[18];
    float* out = (float*)d_out;

    float* ws    = (float*)d_ws;
    float* h     = ws;                 // 1024*256
    float* x1    = ws + 262144;        // 1024*512
    float* tb    = ws + 786432;        // 1024*256
    float* sb    = ws + 1048576;       // 1024*256
    float* nb    = ws + 1310720;       // 1024*256
    float* hi    = ws + 1572864;       // 1024*256
    float* hj    = ws + 1835008;       // 1024*256
    float* zbias = ws + 2097152;       // 512 zeros

    dim3 blk(256);
    zero_fill<<<dim3(2), blk, 0, stream>>>(zbias, 512);

    // FeatureEncoder
    gemm_f32_wmma<1><<<dim3(32, 8, 1), blk, 0, stream>>>(emb, 640, 0, ew1, 512, 0, eb1,
                                                         x1, 512, 0, 640);
    gemm_f32_wmma<0><<<dim3(16, 8, 1), blk, 0, stream>>>(x1, 512, 0, ew2, 256, 0, eb2,
                                                         h, 256, 0, 512);
    // GCN layers
    for (int l = 0; l < 3; ++l) {
        gemm_f32_wmma<0><<<dim3(16, 8, 1), blk, 0, stream>>>(h, 256, 0, wedge + l * 65536, 256, 0,
                                                             bedge + l * 256, tb, 256, 0, 256);
        gemm_f32_wmma<0><<<dim3(16, 8, 1), blk, 0, stream>>>(h, 256, 0, wself + l * 65536, 256, 0,
                                                             bself + l * 256, sb, 256, 0, 256);
        // neigh = bp @ t, batched over B
        gemm_f32_wmma<0><<<dim3(16, 2, 4), blk, 0, stream>>>(bp, 256, 65536, tb, 256, 65536,
                                                             zbias, nb, 256, 65536, 256);
        relu_add_ln<<<dim3(1024), blk, 0, stream>>>(sb, nb, lng + l * 256, lnb + l * 256, h);
    }
    // hi / hj projections (b1 folded into fused scorer)
    gemm_f32_wmma<0><<<dim3(16, 8, 1), blk, 0, stream>>>(h, 256, 0, mw1, 256, 0,
                                                         zbias, hi, 256, 0, 256);
    gemm_f32_wmma<0><<<dim3(16, 8, 1), blk, 0, stream>>>(h, 256, 0, mw1 + 256 * 256, 256, 0,
                                                         zbias, hj, 256, 0, 256);
    // Fused pairwise scorer (dominant 17.2 GFLOP, zero intermediate traffic)
    pairwise_score_wmma<<<dim3(16, 16, 4), blk, 0, stream>>>(hi, hj, bp, mw1, mb1,
                                                             mw2, mb2, mw3, mb3, out);
}
